// TGAT_44598940402287
// MI455X (gfx1250) — matrix operand order repaired and checked by reading, not compiled, not run on hardware
//
#include <hip/hip_runtime.h>
#include <hip/hip_bf16.h>

typedef __attribute__((ext_vector_type(16))) _Float16 v16h;
typedef __attribute__((ext_vector_type(8)))  _Float16 v8h;
typedef __attribute__((ext_vector_type(8)))  float    v8f;

// ---------------------------------------------------------------------------
// WMMA f16 16x16x32, f32 accum. Layouts per CDNA5 ISA:
//   A (16x32 f16): lanes 0-15 row M=lane,    halves 0-7 -> K=0..7,  8-15 -> K=16..23
//                  lanes 16-31 row M=lane-16, halves 0-7 -> K=8..15, 8-15 -> K=24..31
//   B (32x16 f16): lanes 0-15 col N=lane, halves 0-7 -> K=0..7, 8-15 -> K=16..23; lanes 16-31 K=8..15, 24..31
//   C/D (16x16 f32): VGPR r: lanes 0-15 -> (M=r, N=lane); lanes 16-31 -> (M=8+r)
// ---------------------------------------------------------------------------

__device__ __forceinline__ v8f wmma_f16(v16h a, v16h b, v8f c) {
  return __builtin_amdgcn_wmma_f32_16x16x32_f16(false, a, false, b,
                                                (short)0, c, false, false);
}

// Wave-uniform value -> SGPR (scalar branches / scalar addressing).
__device__ __forceinline__ int uni(int v) {
  return __builtin_amdgcn_readfirstlane(v);
}

// Slow-path B fragment builder (used only by the one-shot pack kernel).
__device__ __forceinline__ v16h ldB_pad(const float* B, int ldb, int k0,
                                        int kvalid, int n0, int lane) {
  int n  = n0 + (lane & 15);
  int kb = (lane & 16) ? 8 : 0;
  v16h b;
#pragma unroll
  for (int i = 0; i < 16; ++i) {
    int k = kb + i + ((i & 8) ? 8 : 0);
    b[i] = (k < kvalid) ? (_Float16)B[(long)(k0 + k) * ldb + n] : (_Float16)0.f;
  }
  return b;
}

// Fast fragment fetch from pre-packed buffer: [frag][lane][16 halves].
__device__ __forceinline__ v16h ldFrag(const _Float16* __restrict__ P, int f, int lane) {
  return *(const v16h*)(P + (size_t)f * 512 + lane * 16);
}

// A fragment from f16 row-major matrix; runs of 8 halves load as b128.
// kvalid must be a multiple of 8 (run granularity zero-padding).
__device__ __forceinline__ v16h ldA_f16(const _Float16* __restrict__ A, int lda,
                                        int row0, int k0, int kvalid, int lane,
                                        int maxRow) {
  int m = row0 + (lane & 15);
  if (m > maxRow) m = maxRow;
  int kb = (lane & 16) ? 8 : 0;
  const _Float16* p = A + (size_t)m * lda + k0;
  v16h a;
  if (kb < kvalid) {
    v8h lo = *(const v8h*)(p + kb);
#pragma unroll
    for (int i = 0; i < 8; ++i) a[i] = lo[i];
  } else {
#pragma unroll
    for (int i = 0; i < 8; ++i) a[i] = (_Float16)0.f;
  }
  if (16 + kb < kvalid) {
    v8h hi = *(const v8h*)(p + 16 + kb);
#pragma unroll
    for (int i = 0; i < 8; ++i) a[i + 8] = hi[i];
  } else {
#pragma unroll
    for (int i = 0; i < 8; ++i) a[i + 8] = (_Float16)0.f;
  }
  return a;
}

// ---- ordered-uint encoding so float max can use integer atomicMax ----------
__device__ __forceinline__ unsigned ordered_of_float(float f) {
  unsigned u = __float_as_uint(f);
  return (u & 0x80000000u) ? ~u : (u | 0x80000000u);
}
__device__ __forceinline__ float float_of_ordered(unsigned u) {
  u = (u & 0x80000000u) ? (u ^ 0x80000000u) : ~u;
  return __uint_as_float(u);
}
#define ORDERED_NEG_INF 0x007FFFFFu

// Fragment table ids in the packed weight buffer:
//  0..7  Wlin (kc 0..3) x (nt 0..1)
//  8..9  Wcomb rows 0..31, nt 0..1
// 10..11 Wcomb rows 32..39 (zero padded), nt 0..1
// 12..13 Wq | 14..15 Wk | 16..17 Wv | 18..19 Wskip | 20..21 We
#define NFRAG 22

// ===========================================================================
// One-shot pack: weight fragments -> f16 frag layout; (Wt,bt) per-lane table
// ===========================================================================
__global__ void k_pack(const float* __restrict__ Wlin, const float* __restrict__ Wcomb,
                       const float* __restrict__ Wq, const float* __restrict__ Wk,
                       const float* __restrict__ Wv, const float* __restrict__ Ws,
                       const float* __restrict__ We, const float* __restrict__ Wt,
                       const float* __restrict__ bt,
                       _Float16* __restrict__ packed, float* __restrict__ wtbt) {
  int wave = uni((int)(blockIdx.x * blockDim.x + threadIdx.x) >> 5);
  int lane = threadIdx.x & 31;
  if (wave < NFRAG) {
    const float* W; int k0 = 0, kv = 32, n0;
    int f = wave;
    if (f < 8)       { W = Wlin;  k0 = (f >> 1) * 32; n0 = (f & 1) * 16; }
    else if (f < 10) { W = Wcomb; n0 = (f - 8) * 16; }
    else if (f < 12) { W = Wcomb; k0 = 32; kv = 8; n0 = (f - 10) * 16; }
    else if (f < 14) { W = Wq;    n0 = (f - 12) * 16; }
    else if (f < 16) { W = Wk;    n0 = (f - 14) * 16; }
    else if (f < 18) { W = Wv;    n0 = (f - 16) * 16; }
    else if (f < 20) { W = Ws;    n0 = (f - 18) * 16; }
    else             { W = We;    n0 = (f - 20) * 16; }
    v16h b = ldB_pad(W, 32, k0, kv, n0, lane);
    *(v16h*)(packed + (size_t)f * 512 + lane * 16) = b;
  } else if (wave == NFRAG) {
    // per-lane (Wt[j], bt[j]) pairs in A-fragment K order
    int kb = (lane & 16) ? 8 : 0;
#pragma unroll
    for (int i = 0; i < 8; ++i) {
      int j0 = kb + i, j1 = 16 + kb + i;
      wtbt[lane * 32 + i * 2 + 0]      = Wt[j0];
      wtbt[lane * 32 + i * 2 + 1]      = bt[j0];
      wtbt[lane * 32 + 16 + i * 2 + 0] = Wt[j1];
      wtbt[lane * 32 + 16 + i * 2 + 1] = bt[j1];
    }
  }
}

// ===========================================================================
// x (f32) -> xh (f16)
// ===========================================================================
__global__ void k_xcvt(const float* __restrict__ x, _Float16* __restrict__ xh, int n) {
  int i = blockIdx.x * blockDim.x + threadIdx.x;
  if (i < n) xh[i] = (_Float16)x[i];
}

// ===========================================================================
// init segment max / sum buffers
// ===========================================================================
__global__ void k_init(unsigned* __restrict__ amax, float* __restrict__ asum, int n2) {
  int i = blockIdx.x * blockDim.x + threadIdx.x;
  if (i < n2) { amax[i] = ORDERED_NEG_INF; asum[i] = 0.f; }
}

// ===========================================================================
// h1cat[N,40] (f16) = [relu(x@Wlin+blin) | relu(deg@Wdeg+bdeg)]
// ===========================================================================
__global__ void k_node_lin(const _Float16* __restrict__ xh, const float* __restrict__ ndeg,
                           const _Float16* __restrict__ packed,
                           const float* __restrict__ blin,
                           const float* __restrict__ Wdeg, const float* __restrict__ bdeg,
                           _Float16* __restrict__ h1cat, int N) {
  int wave = uni((int)(blockIdx.x * blockDim.x + threadIdx.x) >> 5);
  int lane = threadIdx.x & 31;
  int tiles = (N + 15) >> 4;
  if (wave >= tiles) return;   // scalar branch
  int row0 = wave << 4;

  v8f acc[2];
  acc[0] = (v8f){0.f,0.f,0.f,0.f,0.f,0.f,0.f,0.f};
  acc[1] = acc[0];
#pragma unroll
  for (int kc = 0; kc < 4; ++kc) {
    v16h a = ldA_f16(xh, 128, row0, kc * 32, 32, lane, N - 1);
    acc[0] = wmma_f16(a, ldFrag(packed, kc * 2 + 0, lane), acc[0]);
    acc[1] = wmma_f16(a, ldFrag(packed, kc * 2 + 1, lane), acc[1]);
  }

  int n = lane & 15, mb = (lane & 16) ? 8 : 0;
  // hoisted invariants (loaded unconditionally, outside divergent stores)
  float b0 = blin[n], b1 = blin[16 + n];
  int mrow = row0 + (lane & 15);
  int mrowc = (mrow < N) ? mrow : N - 1;
  float nd = ndeg[mrowc];
  int cb = (lane & 16) ? 4 : 0;
  float wd0 = Wdeg[cb + 0], wd1 = Wdeg[cb + 1], wd2 = Wdeg[cb + 2], wd3 = Wdeg[cb + 3];
  float bd0 = bdeg[cb + 0], bd1 = bdeg[cb + 1], bd2 = bdeg[cb + 2], bd3 = bdeg[cb + 3];

  if (row0 + 16 <= N) {   // scalar condition -> straight-line fast path
#pragma unroll
    for (int r = 0; r < 8; ++r) {
      size_t rowb = (size_t)(row0 + mb + r) * 40;
      h1cat[rowb + n]      = (_Float16)fmaxf(acc[0][r] + b0, 0.f);
      h1cat[rowb + 16 + n] = (_Float16)fmaxf(acc[1][r] + b1, 0.f);
    }
    size_t rowb = (size_t)mrow * 40 + 32 + cb;
    h1cat[rowb + 0] = (_Float16)fmaxf(nd * wd0 + bd0, 0.f);
    h1cat[rowb + 1] = (_Float16)fmaxf(nd * wd1 + bd1, 0.f);
    h1cat[rowb + 2] = (_Float16)fmaxf(nd * wd2 + bd2, 0.f);
    h1cat[rowb + 3] = (_Float16)fmaxf(nd * wd3 + bd3, 0.f);
  } else {
#pragma unroll
    for (int r = 0; r < 8; ++r) {
      int m = row0 + mb + r;
      if (m < N) {
        size_t rowb = (size_t)m * 40;
        h1cat[rowb + n]      = (_Float16)fmaxf(acc[0][r] + b0, 0.f);
        h1cat[rowb + 16 + n] = (_Float16)fmaxf(acc[1][r] + b1, 0.f);
      }
    }
    if (mrow < N) {
      size_t rowb = (size_t)mrow * 40 + 32 + cb;
      h1cat[rowb + 0] = (_Float16)fmaxf(nd * wd0 + bd0, 0.f);
      h1cat[rowb + 1] = (_Float16)fmaxf(nd * wd1 + bd1, 0.f);
      h1cat[rowb + 2] = (_Float16)fmaxf(nd * wd2 + bd2, 0.f);
      h1cat[rowb + 3] = (_Float16)fmaxf(nd * wd3 + bd3, 0.f);
    }
  }
}

// ===========================================================================
// h1[N,32] (f16) = h1cat @ Wcomb + bcomb   (K=40 zero-padded to 64)
// ===========================================================================
__global__ void k_node_comb(const _Float16* __restrict__ h1cat,
                            const _Float16* __restrict__ packed,
                            const float* __restrict__ bcomb,
                            _Float16* __restrict__ h1, int N) {
  int wave = uni((int)(blockIdx.x * blockDim.x + threadIdx.x) >> 5);
  int lane = threadIdx.x & 31;
  int tiles = (N + 15) >> 4;
  if (wave >= tiles) return;
  int row0 = wave << 4;

  v16h a0 = ldA_f16(h1cat, 40, row0, 0, 32, lane, N - 1);
  v16h a1 = ldA_f16(h1cat, 40, row0, 32, 8, lane, N - 1);
  v8f z = (v8f){0.f,0.f,0.f,0.f,0.f,0.f,0.f,0.f};
  int n = lane & 15, mb = (lane & 16) ? 8 : 0;
  float bc0 = bcomb[n], bc1 = bcomb[16 + n];

  v8f acc0 = wmma_f16(a1, ldFrag(packed, 10, lane),
                      wmma_f16(a0, ldFrag(packed, 8, lane), z));
  v8f acc1 = wmma_f16(a1, ldFrag(packed, 11, lane),
                      wmma_f16(a0, ldFrag(packed, 9, lane), z));

  if (row0 + 16 <= N) {
#pragma unroll
    for (int r = 0; r < 8; ++r) {
      size_t rowb = (size_t)(row0 + mb + r) * 32;
      h1[rowb + n]      = (_Float16)(acc0[r] + bc0);
      h1[rowb + 16 + n] = (_Float16)(acc1[r] + bc1);
    }
  } else {
#pragma unroll
    for (int r = 0; r < 8; ++r) {
      int m = row0 + mb + r;
      if (m < N) {
        size_t rowb = (size_t)m * 32;
        h1[rowb + n]      = (_Float16)(acc0[r] + bc0);
        h1[rowb + 16 + n] = (_Float16)(acc1[r] + bc1);
      }
    }
  }
}

// ===========================================================================
// q/k/v (f32) = h1@W* + b*;  conv = h1@Wskip + bskip (skip/root term)
// q is pre-scaled by 1/sqrt(C)=0.25 so the edge kernel skips the multiply.
// ===========================================================================
__global__ void k_node_qkv(const _Float16* __restrict__ h1,
                           const _Float16* __restrict__ packed,
                           const float* __restrict__ bq, const float* __restrict__ bk,
                           const float* __restrict__ bv, const float* __restrict__ bs,
                           float* __restrict__ qb, float* __restrict__ kb,
                           float* __restrict__ vb, float* __restrict__ conv, int N) {
  int wave = uni((int)(blockIdx.x * blockDim.x + threadIdx.x) >> 5);
  int lane = threadIdx.x & 31;
  int tiles = (N + 15) >> 4;
  if (wave >= tiles) return;
  int row0 = wave << 4;

  v16h a = ldA_f16(h1, 32, row0, 0, 32, lane, N - 1);
  v8f z = (v8f){0.f,0.f,0.f,0.f,0.f,0.f,0.f,0.f};
  int n = lane & 15, mb = (lane & 16) ? 8 : 0;
  bool full = (row0 + 16 <= N);   // scalar

  const float* B[4] = {bq, bk, bv, bs};
  float* O[4]       = {qb, kb, vb, conv};
  const float scl[4] = {0.25f, 1.f, 1.f, 1.f};
#pragma unroll
  for (int t = 0; t < 4; ++t) {
#pragma unroll
    for (int nt = 0; nt < 2; ++nt) {
      float bias = B[t][nt * 16 + n];
      v8f acc = wmma_f16(a, ldFrag(packed, 12 + t * 2 + nt, lane), z);
      if (full) {
#pragma unroll
        for (int r = 0; r < 8; ++r)
          O[t][(size_t)(row0 + mb + r) * 32 + nt * 16 + n] = (acc[r] + bias) * scl[t];
      } else {
#pragma unroll
        for (int r = 0; r < 8; ++r) {
          int m = row0 + mb + r;
          if (m < N) O[t][(size_t)m * 32 + nt * 16 + n] = (acc[r] + bias) * scl[t];
        }
      }
    }
  }
}

// ---------------------------------------------------------------------------
// Time-encoding A fragment from per-lane packed (Wt,bt) table.
// ---------------------------------------------------------------------------
__device__ __forceinline__ v16h enc_frag(float rt, const float* __restrict__ wtbt,
                                         int lane) {
  const float* wb = wtbt + lane * 32;
  v16h a;
#pragma unroll
  for (int i = 0; i < 8; ++i) {
    a[i]     = (_Float16)__cosf(rt * wb[i * 2]      + wb[i * 2 + 1]);
    a[i + 8] = (_Float16)__cosf(rt * wb[16 + i * 2] + wb[16 + i * 2 + 1]);
  }
  return a;
}

// ===========================================================================
// alpha[e,h] = <q[dst], k[src]+e> (q pre-scaled); segment max (ordered atomicMax)
// ===========================================================================
template <bool FULL>
__device__ __forceinline__ void edge_alpha_body(
    int e0, int lane, int sm, int dm, const v8f* et,
    const float* __restrict__ qb, const float* __restrict__ kb,
    float* __restrict__ alpha, unsigned* __restrict__ amax, int E) {
  int c = lane & 15, half = (lane & 16) ? 8 : 0;
#pragma unroll
  for (int r = 0; r < 8; ++r) {
    int ei = e0 + half + r;
    bool valid = FULL || (ei < E);
    int s = __shfl(sm, half + r, 32);
    int d = __shfl(dm, half + r, 32);
#pragma unroll
    for (int h = 0; h < 2; ++h) {
      float t = (kb[(size_t)s * 32 + h * 16 + c] + et[h][r]) *
                qb[(size_t)d * 32 + h * 16 + c];
#pragma unroll
      for (int off = 1; off < 16; off <<= 1) t += __shfl_xor(t, off, 32);
      if (c == 0 && valid) {
        alpha[(size_t)ei * 2 + h] = t;
        atomicMax(&amax[(size_t)d * 2 + h], ordered_of_float(t));
      }
    }
  }
}

__global__ void k_edge_alpha(const int* __restrict__ src, const int* __restrict__ dst,
                             const float* __restrict__ node_time,
                             const float* __restrict__ edge_time,
                             const float* __restrict__ wtbt,
                             const _Float16* __restrict__ packed,
                             const float* __restrict__ qb, const float* __restrict__ kb,
                             float* __restrict__ alpha, unsigned* __restrict__ amax, int E) {
  int wave = uni((int)(blockIdx.x * blockDim.x + threadIdx.x) >> 5);
  int lane = threadIdx.x & 31;
  int tiles = (E + 15) >> 4;
  if (wave >= tiles) return;
  int e0 = wave << 4;

  int m = e0 + (lane & 15);
  if (m >= E) m = E - 1;
  int sm = src[m], dm = dst[m];
  float rt = node_time[sm] - edge_time[m];

  v16h a = enc_frag(rt, wtbt, lane);
  v8f z = (v8f){0.f,0.f,0.f,0.f,0.f,0.f,0.f,0.f};
  v8f et[2];
  et[0] = wmma_f16(a, ldFrag(packed, 20, lane), z);   // head 0 channels
  et[1] = wmma_f16(a, ldFrag(packed, 21, lane), z);   // head 1 channels

  if (e0 + 16 <= E)   // scalar branch
    edge_alpha_body<true>(e0, lane, sm, dm, et, qb, kb, alpha, amax, E);
  else
    edge_alpha_body<false>(e0, lane, sm, dm, et, qb, kb, alpha, amax, E);
}

// ===========================================================================
// ea = exp(alpha - amax[dst]); segment sum
// ===========================================================================
__global__ void k_edge_exp(const int* __restrict__ dst, float* __restrict__ alpha,
                           const unsigned* __restrict__ amax, float* __restrict__ asum,
                           int E) {
  int i = blockIdx.x * blockDim.x + threadIdx.x;
  if (i >= 2 * E) return;
  int e = i >> 1, h = i & 1;
  int d = dst[e];
  float mx = float_of_ordered(amax[(size_t)d * 2 + h]);
  float ea = __expf(alpha[i] - mx);
  alpha[i] = ea;
  atomicAdd(&asum[(size_t)d * 2 + h], ea);
}

// ===========================================================================
// asum -> 1/(asum + 1e-16)   (so the edge message pass multiplies, no rcp)
// ===========================================================================
__global__ void k_rinv(float* __restrict__ asum, int n2) {
  int i = blockIdx.x * blockDim.x + threadIdx.x;
  if (i < n2) asum[i] = 1.f / (asum[i] + 1e-16f);
}

// ===========================================================================
// conv[dst] += (v[src]+e) * ea * rsum[dst]   (e recomputed via WMMA)
// ===========================================================================
template <bool FULL>
__device__ __forceinline__ void edge_msg_body(
    int e0, int lane, int sm, int dm, const v8f* et,
    const float* __restrict__ vb, const float* __restrict__ ea,
    const float* __restrict__ rsum, float* __restrict__ conv, int E) {
  int c = lane & 15, half = (lane & 16) ? 8 : 0;
#pragma unroll
  for (int r = 0; r < 8; ++r) {
    int ei = e0 + half + r;
    bool valid = FULL || (ei < E);
    int eic = FULL ? ei : (valid ? ei : E - 1);
    int s = __shfl(sm, half + r, 32);
    int d = __shfl(dm, half + r, 32);
#pragma unroll
    for (int h = 0; h < 2; ++h) {
      float attn = ea[(size_t)eic * 2 + h] * rsum[(size_t)d * 2 + h];
      float val  = (vb[(size_t)s * 32 + h * 16 + c] + et[h][r]) * attn;
      if (valid) atomicAdd(&conv[(size_t)d * 32 + h * 16 + c], val);
    }
  }
}

__global__ void k_edge_msg(const int* __restrict__ src, const int* __restrict__ dst,
                           const float* __restrict__ node_time,
                           const float* __restrict__ edge_time,
                           const float* __restrict__ wtbt,
                           const _Float16* __restrict__ packed,
                           const float* __restrict__ vb, const float* __restrict__ ea,
                           const float* __restrict__ rsum, float* __restrict__ conv,
                           int E) {
  int wave = uni((int)(blockIdx.x * blockDim.x + threadIdx.x) >> 5);
  int lane = threadIdx.x & 31;
  int tiles = (E + 15) >> 4;
  if (wave >= tiles) return;
  int e0 = wave << 4;

  int m = e0 + (lane & 15);
  if (m >= E) m = E - 1;
  int sm = src[m], dm = dst[m];
  float rt = node_time[sm] - edge_time[m];

  v16h a = enc_frag(rt, wtbt, lane);
  v8f z = (v8f){0.f,0.f,0.f,0.f,0.f,0.f,0.f,0.f};
  v8f et[2];
  et[0] = wmma_f16(a, ldFrag(packed, 20, lane), z);
  et[1] = wmma_f16(a, ldFrag(packed, 21, lane), z);

  if (e0 + 16 <= E)
    edge_msg_body<true>(e0, lane, sm, dm, et, vb, ea, rsum, conv, E);
  else
    edge_msg_body<false>(e0, lane, sm, dm, et, vb, ea, rsum, conv, E);
}

// ===========================================================================
// logits = conv @ Wout + bout; log_softmax (2 classes)
// ===========================================================================
__global__ void k_out(const float* __restrict__ conv, const float* __restrict__ Wout,
                      const float* __restrict__ bout, float* __restrict__ out, int N) {
  int n = blockIdx.x * blockDim.x + threadIdx.x;
  if (n >= N) return;
  float l0 = bout[0], l1 = bout[1];
  const float4* row = (const float4*)(conv + (size_t)n * 32);
#pragma unroll
  for (int jq = 0; jq < 8; ++jq) {
    float4 h = row[jq];
    int j = jq * 4;
    l0 += h.x * Wout[(j + 0) * 2] + h.y * Wout[(j + 1) * 2] +
          h.z * Wout[(j + 2) * 2] + h.w * Wout[(j + 3) * 2];
    l1 += h.x * Wout[(j + 0) * 2 + 1] + h.y * Wout[(j + 1) * 2 + 1] +
          h.z * Wout[(j + 2) * 2 + 1] + h.w * Wout[(j + 3) * 2 + 1];
  }
  float mx  = fmaxf(l0, l1);
  float lse = mx + __logf(__expf(l0 - mx) + __expf(l1 - mx));
  out[(size_t)n * 2 + 0] = l0 - lse;
  out[(size_t)n * 2 + 1] = l1 - lse;
}

// ===========================================================================
extern "C" void kernel_launch(void* const* d_in, const int* in_sizes, int n_in,
                              void* d_out, int out_size, void* d_ws, size_t ws_size,
                              hipStream_t stream) {
  const float* x    = (const float*)d_in[0];
  const int*   ei   = (const int*)d_in[1];
  const float* ntme = (const float*)d_in[2];
  const float* etme = (const float*)d_in[3];
  const float* ndeg = (const float*)d_in[4];
  const float* Wt   = (const float*)d_in[5];
  const float* bt   = (const float*)d_in[6];
  const float* Wlin = (const float*)d_in[7];
  const float* blin = (const float*)d_in[8];
  const float* Wdeg = (const float*)d_in[9];
  const float* bdeg = (const float*)d_in[10];
  const float* Wcomb= (const float*)d_in[11];
  const float* bcomb= (const float*)d_in[12];
  const float* Wq   = (const float*)d_in[13];
  const float* bq   = (const float*)d_in[14];
  const float* Wk   = (const float*)d_in[15];
  const float* bk   = (const float*)d_in[16];
  const float* Wv   = (const float*)d_in[17];
  const float* bv   = (const float*)d_in[18];
  const float* We   = (const float*)d_in[19];
  const float* Ws   = (const float*)d_in[20];
  const float* bs   = (const float*)d_in[21];
  const float* Wout = (const float*)d_in[22];
  const float* bout = (const float*)d_in[23];

  const int N = in_sizes[0] / 128;
  const int E = in_sizes[1] / 2;
  const int* src = ei;
  const int* dst = ei + E;

  // --- workspace carving (aligned) ---
  char* base = (char*)d_ws;
  size_t off = 0;
  auto alloc = [&](size_t bytes, size_t align) -> char* {
    off = (off + align - 1) & ~(align - 1);
    char* p = base + off;
    off += bytes;
    return p;
  };
  float* qb    = (float*)alloc((size_t)N * 32 * 4, 256);
  float* kb    = (float*)alloc((size_t)N * 32 * 4, 256);
  float* vb    = (float*)alloc((size_t)N * 32 * 4, 256);
  float* conv  = (float*)alloc((size_t)N * 32 * 4, 256);
  float* alpha = (float*)alloc((size_t)E * 2 * 4, 256);
  unsigned* amax = (unsigned*)alloc((size_t)N * 2 * 4, 256);
  float* asum  = (float*)alloc((size_t)N * 2 * 4, 256);
  float* wtbt  = (float*)alloc((size_t)32 * 32 * 4, 256);
  _Float16* packed = (_Float16*)alloc((size_t)NFRAG * 512 * 2, 256);
  _Float16* xh     = (_Float16*)alloc((size_t)N * 128 * 2, 256);
  _Float16* h1cat  = (_Float16*)alloc((size_t)N * 40 * 2, 256);
  _Float16* h1     = (_Float16*)alloc((size_t)N * 32 * 2, 256);

  const int BLK = 256;  // 8 waves
  int tilesN = (N + 15) >> 4;
  int tilesE = (E + 15) >> 4;
  int gN = (tilesN * 32 + BLK - 1) / BLK;
  int gE = (tilesE * 32 + BLK - 1) / BLK;

  k_pack<<<((NFRAG + 1) * 32 + BLK - 1) / BLK, BLK, 0, stream>>>(
      Wlin, Wcomb, Wq, Wk, Wv, Ws, We, Wt, bt, packed, wtbt);
  k_xcvt<<<((size_t)N * 128 + BLK - 1) / BLK, BLK, 0, stream>>>(x, xh, N * 128);
  k_init<<<(2 * N + BLK - 1) / BLK, BLK, 0, stream>>>(amax, asum, 2 * N);

  k_node_lin<<<gN, BLK, 0, stream>>>(xh, ndeg, packed, blin, Wdeg, bdeg, h1cat, N);
  k_node_comb<<<gN, BLK, 0, stream>>>(h1cat, packed, bcomb, h1, N);
  k_node_qkv<<<gN, BLK, 0, stream>>>(h1, packed, bq, bk, bv, bs, qb, kb, vb, conv, N);

  k_edge_alpha<<<gE, BLK, 0, stream>>>(src, dst, ntme, etme, wtbt, packed,
                                       qb, kb, alpha, amax, E);
  k_edge_exp<<<(2 * E + BLK - 1) / BLK, BLK, 0, stream>>>(dst, alpha, amax, asum, E);
  k_rinv<<<(2 * N + BLK - 1) / BLK, BLK, 0, stream>>>(asum, 2 * N);
  k_edge_msg<<<gE, BLK, 0, stream>>>(src, dst, ntme, etme, wtbt, packed,
                                     vb, alpha, asum, conv, E);
  k_out<<<(N + BLK - 1) / BLK, BLK, 0, stream>>>(conv, Wout, bout, (float*)d_out, N);
}
